// MultiHeadSelfAttention_13357348290568
// MI455X (gfx1250) — matrix operand order repaired and checked by reading, not compiled, not run on hardware
//
#include <hip/hip_runtime.h>
#include <hip/hip_bf16.h>

// MI455X / gfx1250: wave32, WMMA 16x16x32 bf16, async global->LDS staging,
// software-pipelined LDS fragment reads. MHA: B=8, S=1024, D=768, H=12, DK=64.

typedef __bf16 bf16;
typedef __attribute__((ext_vector_type(16))) __bf16 v16bf;
typedef __attribute__((ext_vector_type(8)))  __bf16 v8bf;
typedef __attribute__((ext_vector_type(8)))  float  v8f;

static constexpr int Bd  = 8;
static constexpr int Sd  = 1024;
static constexpr int Dd  = 768;
static constexpr int Hd  = 12;
static constexpr int DKd = 64;

__device__ __forceinline__ v8f v8f_zero() {
    v8f r = {0.f, 0.f, 0.f, 0.f, 0.f, 0.f, 0.f, 0.f};
    return r;
}

__device__ __forceinline__ v16bf cat16(v8bf lo, v8bf hi) {
    return __builtin_shufflevector(lo, hi, 0,1,2,3,4,5,6,7,8,9,10,11,12,13,14,15);
}

// A fragment (16x32, MxK), row-major source with leading dim ld.
// ISA layout: lanes 0-15 -> rows; elems 0-7 = K 8*half.. ; elems 8-15 = K 16+8*half..
__device__ __forceinline__ v16bf load_frag_A(const bf16* base, int ld, int lane) {
    int r  = lane & 15;
    int hl = (lane >> 4) & 1;
    const bf16* p = base + (size_t)r * ld + 8 * hl;
    v8bf lo = *(const v8bf*)(p);
    v8bf hi = *(const v8bf*)(p + 16);
    return cat16(lo, hi);
}

// B fragment (32x16, KxN) from TRANSPOSED storage baseT[n][k] with leading dim ld.
// ISA layout: lanes 0-15 -> cols; elems 0-7 = K 8*half.. ; elems 8-15 = K 16+8*half..
__device__ __forceinline__ v16bf load_frag_BT(const bf16* baseT, int ld, int lane) {
    int n  = lane & 15;
    int hl = (lane >> 4) & 1;
    const bf16* p = baseT + (size_t)n * ld + 8 * hl;
    v8bf lo = *(const v8bf*)(p);
    v8bf hi = *(const v8bf*)(p + 16);
    return cat16(lo, hi);
}

__device__ __forceinline__ v8f wmma_bf16(v16bf a, v16bf b, v8f c) {
    return __builtin_amdgcn_wmma_f32_16x16x32_bf16(false, a, false, b, (short)0, c, false, false);
}

// Async copy of nrows x 32 bf16 (64B rows, stride ld elems) from global into
// contiguous LDS [nrows][32]. 16B per lane-op via global_load_async_to_lds_b128
// (ASYNCcnt-tracked). Caller must s_wait_asynccnt + barrier before use.
__device__ __forceinline__ void async_copy_rows(bf16* lds_dst, const bf16* gsrc,
                                                int ld, int nrows) {
    int t = threadIdx.x;                 // blockDim.x == 128
    int total = nrows * 4;               // b128 transfers
    for (int idx = t; idx < total; idx += 128) {
        int row = idx >> 2, seg = idx & 3;
        unsigned int lds32 =
            (unsigned int)(size_t)(lds_dst + (size_t)idx * 8);   // low 32 bits = LDS offset
        unsigned long long g64 =
            (unsigned long long)(size_t)(gsrc + (size_t)row * ld + seg * 8);
        asm volatile("global_load_async_to_lds_b128 %0, %1, off"
                     :: "v"(lds32), "v"(g64) : "memory");
    }
}

// ---------------- conversion kernels ----------------

__global__ void cvt_bf16_kernel(const float* __restrict__ in, bf16* __restrict__ out, int n) {
    int i = blockIdx.x * blockDim.x + threadIdx.x;
    if (i < n) out[i] = (bf16)in[i];
}

// [H, D, DK] f32 -> [H, DK, D] bf16 (per-head transpose)
__global__ void cvt_headw_T_kernel(const float* __restrict__ in, bf16* __restrict__ out) {
    int i = blockIdx.x * blockDim.x + threadIdx.x;
    if (i >= Hd * Dd * DKd) return;
    int h   = i / (Dd * DKd);
    int rem = i - h * (Dd * DKd);
    int d   = rem / DKd;
    int kc  = rem - d * DKd;
    out[(size_t)h * DKd * Dd + (size_t)kc * Dd + d] = (bf16)in[i];
}

// [D, D] f32 -> transposed bf16
__global__ void cvt_W_T_kernel(const float* __restrict__ in, bf16* __restrict__ out) {
    int i = blockIdx.x * blockDim.x + threadIdx.x;
    if (i >= Dd * Dd) return;
    int r = i / Dd, c = i - r * Dd;
    out[(size_t)c * Dd + r] = (bf16)in[i];
}

// ---------------- QKV projection ----------------
// Block = 4 waves, one (b, head, 64-row s-block). Weight K-chunk tiles
// (3 x 64x32 bf16 = 12KB) double-buffered in LDS via async copies. Per chunk
// each wave runs a 12-stream rotating ds_load->wmma pipeline (B frag for
// stream s+1 in flight during WMMA s) with the A fragment double-buffered
// across chunks.
__global__ void qkv_proj_kernel(const bf16* __restrict__ xbf,
                                const bf16* __restrict__ WqT,
                                const bf16* __restrict__ WkT,
                                const bf16* __restrict__ WvT,
                                bf16* __restrict__ q, bf16* __restrict__ k,
                                bf16* __restrict__ vT) {
    __shared__ __align__(16) bf16 wbuf[2][3][64][32];   // 24KB double buffer

    int lane  = threadIdx.x & 31;
    int wslot = (threadIdx.x >> 5) & 3;
    int bid   = blockIdx.x;
    int st4   = bid & 15;                // 16 s-blocks of 64 rows
    int head  = (bid >> 4) % Hd;
    int b     = (bid >> 4) / Hd;
    int st    = st4 * 4 + wslot;

    const bf16* A0 = xbf + (size_t)(b * Sd + st * 16) * Dd;
    const bf16* Wsrc[3] = { WqT + (size_t)head * DKd * Dd,
                            WkT + (size_t)head * DKd * Dd,
                            WvT + (size_t)head * DKd * Dd };

    // acc[s]: stream s = n*3 + mtx (mtx: 0=q,1=k,2=v ; n: 16-col tile)
    v8f acc[12];
#pragma unroll
    for (int s = 0; s < 12; s++) acc[s] = v8f_zero();

    // prologue: chunk 0 copy in flight + A frag for chunk 0
#pragma unroll
    for (int mx = 0; mx < 3; mx++)
        async_copy_rows(&wbuf[0][mx][0][0], Wsrc[mx], Dd, 64);
    v16bf a_cur = load_frag_A(A0, Dd, lane);

    const int NC = Dd / 32;              // 24 chunks
    for (int ci = 0; ci < NC; ci++) {
        int cur = ci & 1;
        bool has_next = (ci + 1 < NC);
        if (has_next) {
#pragma unroll
            for (int mx = 0; mx < 3; mx++)
                async_copy_rows(&wbuf[cur ^ 1][mx][0][0],
                                Wsrc[mx] + (ci + 1) * 32, Dd, 64);
            asm volatile("s_wait_asynccnt 6" ::: "memory");   // chunk ci landed
        } else {
            asm volatile("s_wait_asynccnt 0" ::: "memory");
        }
        __syncthreads();                                      // all waves' copies visible

        // A fragment for next chunk: global latency hidden under 12 WMMAs
        v16bf a_next = a_cur;
        if (has_next) a_next = load_frag_A(A0 + (ci + 1) * 32, Dd, lane);

        // rotating B-fragment pipeline over 12 streams
        v16bf bcur = load_frag_BT(&wbuf[cur][0][0][0], 32, lane);
#pragma unroll
        for (int s = 0; s < 12; s++) {
            v16bf bnext = bcur;
            if (s + 1 < 12) {
                int mtx = (s + 1) % 3, n = (s + 1) / 3;
                bnext = load_frag_BT(&wbuf[cur][mtx][n * 16][0], 32, lane);
            }
            acc[s] = wmma_bf16(a_cur, bcur, acc[s]);
            bcur = bnext;
        }
        __syncthreads();                 // reads done before next async overwrite
        a_cur = a_next;
    }

    int col = lane & 15, hl = (lane >> 4) & 1;
    size_t bh = (size_t)(b * Hd + head);
    bf16* qb = q  + (bh * Sd + st * 16) * DKd;
    bf16* kb = k  + (bh * Sd + st * 16) * DKd;
    bf16* vb = vT + bh * DKd * Sd + st * 16;   // [DK, S] transposed store
#pragma unroll
    for (int n = 0; n < 4; n++) {
#pragma unroll
        for (int j = 0; j < 8; j++) {
            int m = j + 8 * hl;
            qb[(size_t)m * DKd + n * 16 + col] = (bf16)acc[n * 3 + 0][j];
            kb[(size_t)m * DKd + n * 16 + col] = (bf16)acc[n * 3 + 1][j];
            vb[(size_t)(n * 16 + col) * Sd + m] = (bf16)acc[n * 3 + 2][j];
        }
    }
}

// ---------------- flash attention ----------------
// One wave = 16 q rows of one (b, head). Online softmax over S in 32-key tiles.
// K fragments software-pipelined one tile ahead; V fragments issued at top of
// the iteration so their latency overlaps the score WMMAs + softmax VALU work.
__global__ void attn_kernel(const bf16* __restrict__ q, const bf16* __restrict__ k,
                            const bf16* __restrict__ vT, bf16* __restrict__ z) {
    __shared__ __align__(16) bf16 pbuf[4][16][32];   // per-wave P transpose staging
    int lane  = threadIdx.x & 31;
    int wslot = (threadIdx.x >> 5) & 3;
    int w     = (blockIdx.x * blockDim.x + threadIdx.x) >> 5;
    int st    = w & 63;
    int head  = (w >> 6) % Hd;
    int b     = (w >> 6) / Hd;
    size_t bh = (size_t)(b * Hd + head);

    const bf16* qb = q  + (bh * Sd + st * 16) * DKd;
    const bf16* kb = k  + bh * Sd * DKd;
    const bf16* vb = vT + bh * DKd * Sd;

    v16bf qa0 = load_frag_A(qb,      DKd, lane);
    v16bf qa1 = load_frag_A(qb + 32, DKd, lane);

    v8f o[4];
    float m[8], l[8];
#pragma unroll
    for (int n = 0; n < 4; n++) o[n] = v8f_zero();
#pragma unroll
    for (int j = 0; j < 8; j++) { m[j] = -3.0e38f; l[j] = 0.f; }

    const float scale = 0.125f;   // 1/sqrt(64)
    int hl = (lane >> 4) & 1;
    int col = lane & 15;

    // prologue: K fragments for tile 0
    v16bf kf0 = load_frag_BT(kb,                       DKd, lane);
    v16bf kf1 = load_frag_BT(kb + 32,                  DKd, lane);
    v16bf kf2 = load_frag_BT(kb + (size_t)16 * DKd,      DKd, lane);
    v16bf kf3 = load_frag_BT(kb + (size_t)16 * DKd + 32, DKd, lane);

    for (int t0 = 0; t0 < Sd; t0 += 32) {
        // V fragments issued early; consumed only after softmax
        v16bf vf[4];
#pragma unroll
        for (int n = 0; n < 4; n++)
            vf[n] = load_frag_BT(vb + (size_t)(n * 16) * Sd + t0, Sd, lane);

        v8f s0 = v8f_zero(), s1 = v8f_zero();
        s0 = wmma_bf16(qa0, kf0, s0);
        s0 = wmma_bf16(qa1, kf1, s0);
        s1 = wmma_bf16(qa0, kf2, s1);
        s1 = wmma_bf16(qa1, kf3, s1);

        // prefetch K fragments for next tile; latency hidden by softmax below
        if (t0 + 32 < Sd) {
            const bf16* kn = kb + (size_t)(t0 + 32) * DKd;
            kf0 = load_frag_BT(kn,                       DKd, lane);
            kf1 = load_frag_BT(kn + 32,                  DKd, lane);
            kf2 = load_frag_BT(kn + (size_t)16 * DKd,      DKd, lane);
            kf3 = load_frag_BT(kn + (size_t)16 * DKd + 32, DKd, lane);
        }

#pragma unroll
        for (int j = 0; j < 8; j++) {
            float a0 = s0[j] * scale, a1 = s1[j] * scale;
            float vmax = fmaxf(a0, a1);
            vmax = fmaxf(vmax, __shfl_xor(vmax, 1, 32));
            vmax = fmaxf(vmax, __shfl_xor(vmax, 2, 32));
            vmax = fmaxf(vmax, __shfl_xor(vmax, 4, 32));
            vmax = fmaxf(vmax, __shfl_xor(vmax, 8, 32));
            float newm = fmaxf(m[j], vmax);
            float corr = __expf(m[j] - newm);
            float p0 = __expf(a0 - newm);
            float p1 = __expf(a1 - newm);
            float rs = p0 + p1;
            rs += __shfl_xor(rs, 1, 32);
            rs += __shfl_xor(rs, 2, 32);
            rs += __shfl_xor(rs, 4, 32);
            rs += __shfl_xor(rs, 8, 32);
            l[j] = l[j] * corr + rs;
            m[j] = newm;
#pragma unroll
            for (int n = 0; n < 4; n++) o[n][j] *= corr;
            int row = j + 8 * hl;
            pbuf[wslot][row][col]      = (bf16)p0;
            pbuf[wslot][row][col + 16] = (bf16)p1;
        }
        // in-wave C->A transpose through LDS; DS ops are in-order per wave
        asm volatile("s_wait_dscnt 0" ::: "memory");
        v16bf pa = load_frag_A(&pbuf[wslot][0][0], 32, lane);
#pragma unroll
        for (int n = 0; n < 4; n++)
            o[n] = wmma_bf16(pa, vf[n], o[n]);
    }

    // normalize and write z in concat layout [B*S, H*DK]
    bf16* zrow = z + (size_t)(b * Sd + st * 16) * (Hd * DKd) + head * DKd;
#pragma unroll
    for (int j = 0; j < 8; j++) {
        float inv = 1.0f / l[j];
        int mrow = j + 8 * hl;
#pragma unroll
        for (int n = 0; n < 4; n++)
            zrow[(size_t)mrow * (Hd * DKd) + n * 16 + col] = (bf16)(o[n][j] * inv);
    }
}

// ---------------- output projection: out = z @ W ----------------
// Block = 4 waves sharing one 64-col W stripe; stripe K-chunks (64x32 bf16)
// double-buffered in LDS via async copies; rotating ds_load->wmma pipeline and
// A fragment double-buffered across chunks.
__global__ void out_proj_kernel(const bf16* __restrict__ zbf, const bf16* __restrict__ WT,
                                float* __restrict__ out) {
    __shared__ __align__(16) bf16 wbuf[2][64][32];   // 8KB double buffer

    int lane  = threadIdx.x & 31;
    int wslot = (threadIdx.x >> 5) & 3;
    int bid   = blockIdx.x;
    int ct    = bid % (Dd / 64);         // 12 col tiles of 64
    int rtq   = bid / (Dd / 64);         // 128 quads of row tiles
    int rt    = rtq * 4 + wslot;

    const bf16* A0  = zbf + (size_t)rt * 16 * Dd;
    const bf16* BT0 = WT  + (size_t)ct * 64 * Dd;

    v8f acc[4];
#pragma unroll
    for (int n = 0; n < 4; n++) acc[n] = v8f_zero();

    async_copy_rows(&wbuf[0][0][0], BT0, Dd, 64);
    v16bf a_cur = load_frag_A(A0, Dd, lane);

    const int NC = Dd / 32;
    for (int ci = 0; ci < NC; ci++) {
        int cur = ci & 1;
        bool has_next = (ci + 1 < NC);
        if (has_next) {
            async_copy_rows(&wbuf[cur ^ 1][0][0], BT0 + (ci + 1) * 32, Dd, 64);
            asm volatile("s_wait_asynccnt 2" ::: "memory");
        } else {
            asm volatile("s_wait_asynccnt 0" ::: "memory");
        }
        __syncthreads();

        v16bf a_next = a_cur;
        if (has_next) a_next = load_frag_A(A0 + (ci + 1) * 32, Dd, lane);

        v16bf bcur = load_frag_BT(&wbuf[cur][0][0], 32, lane);
#pragma unroll
        for (int n = 0; n < 4; n++) {
            v16bf bnext = bcur;
            if (n + 1 < 4)
                bnext = load_frag_BT(&wbuf[cur][(n + 1) * 16][0], 32, lane);
            acc[n] = wmma_bf16(a_cur, bcur, acc[n]);
            bcur = bnext;
        }
        __syncthreads();
        a_cur = a_next;
    }

    int col = lane & 15, hl = (lane >> 4) & 1;
#pragma unroll
    for (int n = 0; n < 4; n++)
#pragma unroll
        for (int j = 0; j < 8; j++)
            out[(size_t)(rt * 16 + j + 8 * hl) * Dd + ct * 64 + n * 16 + col] = acc[n][j];
}

// ---------------- host launcher ----------------
extern "C" void kernel_launch(void* const* d_in, const int* in_sizes, int n_in,
                              void* d_out, int out_size, void* d_ws, size_t ws_size,
                              hipStream_t stream) {
    (void)in_sizes; (void)n_in; (void)out_size; (void)ws_size;
    const float* x  = (const float*)d_in[0];
    const float* Wq = (const float*)d_in[1];
    const float* Wk = (const float*)d_in[2];
    const float* Wv = (const float*)d_in[3];
    const float* W  = (const float*)d_in[4];
    float* out = (float*)d_out;

    char* ws = (char*)d_ws;
    size_t off = 0;
    auto alloc = [&](size_t bytes) -> void* {
        void* p = ws + off;
        off += (bytes + 255) & ~(size_t)255;
        return p;
    };

    const size_t nX  = (size_t)Bd * Sd * Dd;          // 6.29M
    const size_t nHW = (size_t)Hd * Dd * DKd;         // 0.59M
    const size_t nQ  = (size_t)Bd * Hd * Sd * DKd;    // 6.29M

    bf16* xbf = (bf16*)alloc(nX  * sizeof(bf16));
    bf16* WqT = (bf16*)alloc(nHW * sizeof(bf16));
    bf16* WkT = (bf16*)alloc(nHW * sizeof(bf16));
    bf16* WvT = (bf16*)alloc(nHW * sizeof(bf16));
    bf16* WT  = (bf16*)alloc((size_t)Dd * Dd * sizeof(bf16));
    bf16* qd  = (bf16*)alloc(nQ * sizeof(bf16));
    bf16* kd  = (bf16*)alloc(nQ * sizeof(bf16));
    bf16* vTd = (bf16*)alloc(nQ * sizeof(bf16));
    bf16* zbf = (bf16*)alloc(nX * sizeof(bf16));

    cvt_bf16_kernel<<<(int)((nX + 255) / 256), 256, 0, stream>>>(x, xbf, (int)nX);
    cvt_headw_T_kernel<<<(int)((nHW + 255) / 256), 256, 0, stream>>>(Wq, WqT);
    cvt_headw_T_kernel<<<(int)((nHW + 255) / 256), 256, 0, stream>>>(Wk, WkT);
    cvt_headw_T_kernel<<<(int)((nHW + 255) / 256), 256, 0, stream>>>(Wv, WvT);
    cvt_W_T_kernel<<<(Dd * Dd + 255) / 256, 256, 0, stream>>>(W, WT);

    qkv_proj_kernel<<<1536, 128, 0, stream>>>(xbf, WqT, WkT, WvT, qd, kd, vTd);
    attn_kernel<<<1536, 128, 0, stream>>>(qd, kd, vTd, zbf);
    out_proj_kernel<<<1536, 128, 0, stream>>>(zbf, WT, out);
}
